// MergeAttentionLayer_16982300688807
// MI455X (gfx1250) — matrix operand order repaired and checked by reading, not compiled, not run on hardware
//
#include <hip/hip_runtime.h>
#include <hip/hip_bf16.h>

// ---------------------------------------------------------------------------
// CDNA5 (gfx1250) WMMA types & helpers
// ---------------------------------------------------------------------------
typedef __attribute__((ext_vector_type(16))) __bf16 v16bf;
typedef __attribute__((ext_vector_type(8)))  float  v8f;
typedef __attribute__((ext_vector_type(4)))  unsigned int u32x4;
typedef __attribute__((ext_vector_type(8)))  int   i32x8;
typedef __attribute__((ext_vector_type(4)))  int   i32x4;

union FragU {
  v16bf v;
  uint4 q[2];
};

__device__ __forceinline__ unsigned short f2bf(float x) {
  unsigned int u = __float_as_uint(x);
  u += 0x7FFFu + ((u >> 16) & 1u);           // round-to-nearest-even
  return (unsigned short)(u >> 16);
}

__device__ __forceinline__ v8f vzero8() {
  v8f z = {0.f, 0.f, 0.f, 0.f, 0.f, 0.f, 0.f, 0.f};
  return z;
}

__device__ __forceinline__ v8f wmma_bf16(v16bf a, v16bf b, v8f c) {
  return __builtin_amdgcn_wmma_f32_16x16x32_bf16(
      false, a, false, b, (short)0, c, false, false);
}

// A fragment: 16x32 bf16 tile stored row-major [16][stride] in LDS.
__device__ __forceinline__ v16bf load_frag_a(const unsigned short* p, int stride) {
  const int lane = threadIdx.x & 31;
  const unsigned short* q = p + (lane & 15) * stride + ((lane >> 4) << 3);
  FragU f;
  f.q[0] = *(const uint4*)(q);        // K = kb .. kb+7
  f.q[1] = *(const uint4*)(q + 16);   // K = kb+16 .. kb+23
  return f.v;
}

// B fragment: 32x16 bf16 tile staged TRANSPOSED in LDS as [n][k] (k contiguous).
__device__ __forceinline__ v16bf load_frag_b(const unsigned short* p, int stride) {
  const int lane = threadIdx.x & 31;
  const unsigned short* q = p + (lane & 15) * stride + ((lane >> 4) << 3);
  FragU f;
  f.q[0] = *(const uint4*)(q);        // K = kb .. kb+7
  f.q[1] = *(const uint4*)(q + 16);   // K = kb+16 .. kb+23
  return f.v;
}

// ---------------------------------------------------------------------------
// TDM 2D tile load (bf16): global [rows x cols] strided -> LDS with row pad.
// D# per CDNA5 ISA (group0 128b, group1 256b).
// Issued whole-wave (EXEC ignored); tracked by TENSORcnt.
// NOTE: this toolchain exposes the 6-arg builtin (extra i32x8 group).
// ---------------------------------------------------------------------------
__device__ __forceinline__ void tdm_load_2d_bf16(
    unsigned lds_byte, const void* gptr,
    int tile_x, int tile_y, long row_stride_elems,
    int pad_interval_code, int pad_amount_code) {
  const unsigned long long ga = (unsigned long long)gptr;
  u32x4 g0;
  g0[0] = 1u;                                         // count=1, user desc
  g0[1] = lds_byte;                                   // lds_addr
  g0[2] = (unsigned)(ga & 0xFFFFFFFFu);               // global_addr[31:0]
  g0[3] = (unsigned)((ga >> 32) & 0x1FFFFFFu) | (2u << 30);  // [56:32], type=2
  const unsigned td0 = (unsigned)row_stride_elems;    // tensor_dim0 (x extent)
  const unsigned td1 = 1u << 20;                      // tensor_dim1 (generous)
  const unsigned long long s0 = (unsigned long long)row_stride_elems;
  i32x8 g1;
  g1[0] = (int)((1u << 16)                            // data_size = 2B
              | (1u << 20)                            // pad_enable
              | ((unsigned)pad_interval_code << 22)
              | ((unsigned)pad_amount_code << 25));
  g1[1] = (int)((td0 & 0xFFFFu) << 16);               // dim0[15:0] @ [63:48]
  g1[2] = (int)(((td0 >> 16) & 0xFFFFu) | ((td1 & 0xFFFFu) << 16));
  g1[3] = (int)(((td1 >> 16) & 0xFFFFu) | (((unsigned)tile_x & 0xFFFFu) << 16));
  g1[4] = (int)((unsigned)tile_y & 0xFFFFu);          // tile_dim1; tile_dim2=0
  g1[5] = (int)(unsigned)(s0 & 0xFFFFFFFFull);        // dim0_stride[31:0]
  g1[6] = (int)(unsigned)((s0 >> 32) & 0xFFFFull);    // dim0_stride[47:32]
  g1[7] = 0;
  i32x4 gz4 = {0, 0, 0, 0};
  i32x8 gz8 = {0, 0, 0, 0, 0, 0, 0, 0};
  __builtin_amdgcn_tensor_load_to_lds(g0, g1, gz4, gz4, gz8, 0);
}

__device__ __forceinline__ unsigned lds_off(const void* p) {
  return (unsigned)(unsigned long long)p;   // LDS generic addr: offset in [31:0]
}

// ---------------------------------------------------------------------------
// Tiled GEMM:  C[T,N] = act( A[T,K] x B[K,N] + bias )
//   Block tile 128x128, 256 threads = 8 waves (4 row-strips x 2 col-strips),
//   each wave: 32x64 = 8 x v_wmma_f32_16x16x32_bf16 per k-step.
//   LDS double-buffered (ping-pong): tile t+1 loads overlap tile t WMMAs.
// ---------------------------------------------------------------------------
#define GEMM_STRIDE 40   // 32 + 8 pad (ushorts); 80B rows, 16B-aligned

template <bool A_BF16, bool OUT_BF16, bool RELU>
__global__ __launch_bounds__(256) void gemm_wmma(
    const void* __restrict__ Aopaque, const float* __restrict__ B,
    const float* __restrict__ bias, void* __restrict__ Copaque,
    int T, int N, int K) {
  __shared__ __align__(16) unsigned short As[2][128][GEMM_STRIDE];  // [row][k]
  __shared__ __align__(16) unsigned short Bs[2][128][GEMM_STRIDE];  // [n][k]

  const int tid  = threadIdx.x;
  const int wave = tid >> 5;
  const int lane = tid & 31;
  const int wr   = wave >> 1;   // 0..3  (32-row strip)
  const int wc   = wave & 1;    // 0..1  (64-col strip)
  const int row0 = blockIdx.y * 128;
  const int col0 = blockIdx.x * 128;

  const float*          Af = (const float*)Aopaque;
  const unsigned short* Ab = (const unsigned short*)Aopaque;

  const int a_r = tid >> 1;          // 0..127
  const int a_k = (tid & 1) * 16;    // 0 or 16
  const int b_n = tid & 127;         // 0..127
  const int b_k = (tid >> 7) * 16;   // 0 or 16

  v8f acc[2][4];
#pragma unroll
  for (int i = 0; i < 2; ++i)
#pragma unroll
    for (int j = 0; j < 4; ++j) acc[i][j] = vzero8();

  auto load_tile = [&](int kb, int buf) {
    // ---- A: each thread 16 contiguous k ----
    const long src = (long)(row0 + a_r) * K + kb + a_k;
    if constexpr (A_BF16) {
      *(uint4*)&As[buf][a_r][a_k]     = *(const uint4*)(Ab + src);
      *(uint4*)&As[buf][a_r][a_k + 8] = *(const uint4*)(Ab + src + 8);
    } else {
#pragma unroll
      for (int hh = 0; hh < 2; ++hh) {
        float4 v0 = *(const float4*)(Af + src + hh * 8);
        float4 v1 = *(const float4*)(Af + src + hh * 8 + 4);
        uint4 packed;
        unsigned int* pw = (unsigned int*)&packed;
        pw[0] = (unsigned int)f2bf(v0.x) | ((unsigned int)f2bf(v0.y) << 16);
        pw[1] = (unsigned int)f2bf(v0.z) | ((unsigned int)f2bf(v0.w) << 16);
        pw[2] = (unsigned int)f2bf(v1.x) | ((unsigned int)f2bf(v1.y) << 16);
        pw[3] = (unsigned int)f2bf(v1.z) | ((unsigned int)f2bf(v1.w) << 16);
        *(uint4*)&As[buf][a_r][a_k + hh * 8] = packed;
      }
    }
    if (kb + 64 < K) {   // speculative prefetch of tile t+2 -> global_prefetch_b8
      const char* pf = (const char*)Aopaque + ((src + 32) << (A_BF16 ? 1 : 2));
      __builtin_prefetch(pf, 0, 0);
    }
    // ---- B transposed: Bs[n][k] = B[kb+k][col0+n], 16 strided reads ----
    const float* bp = B + (long)(kb + b_k) * N + col0 + b_n;
#pragma unroll
    for (int hh = 0; hh < 2; ++hh) {
      uint4 packed;
      unsigned int* pw = (unsigned int*)&packed;
#pragma unroll
      for (int i = 0; i < 4; ++i) {
        unsigned int lo = f2bf(bp[(long)(hh * 8 + 2 * i) * N]);
        unsigned int hi = f2bf(bp[(long)(hh * 8 + 2 * i + 1) * N]);
        pw[i] = lo | (hi << 16);
      }
      *(uint4*)&Bs[buf][b_n][b_k + hh * 8] = packed;
    }
  };

  load_tile(0, 0);
  const int nkt = K >> 5;
  for (int t = 0; t < nkt; ++t) {
    __syncthreads();
    if (t + 1 < nkt) load_tile((t + 1) << 5, (t + 1) & 1);
    const int buf = t & 1;
    v16bf a0 = load_frag_a(&As[buf][wr * 32][0],      GEMM_STRIDE);
    v16bf a1 = load_frag_a(&As[buf][wr * 32 + 16][0], GEMM_STRIDE);
#pragma unroll
    for (int j = 0; j < 4; ++j) {
      v16bf bfrag = load_frag_b(&Bs[buf][wc * 64 + j * 16][0], GEMM_STRIDE);
      acc[0][j] = wmma_bf16(a0, bfrag, acc[0][j]);
      acc[1][j] = wmma_bf16(a1, bfrag, acc[1][j]);
    }
  }

  // ---- epilogue: bias (+ReLU), store f32 or bf16 ----
  const int hgrp = lane >> 4;
  const int cn   = lane & 15;
#pragma unroll
  for (int j = 0; j < 4; ++j) {
    const int col = col0 + wc * 64 + j * 16 + cn;
    const float bv = bias[col];
#pragma unroll
    for (int i = 0; i < 2; ++i) {
#pragma unroll
      for (int r = 0; r < 8; ++r) {
        const int row = row0 + wr * 32 + i * 16 + hgrp * 8 + r;
        float v = acc[i][j][r] + bv;
        if (RELU) v = fmaxf(v, 0.f);
        if constexpr (OUT_BF16)
          ((unsigned short*)Copaque)[(long)row * N + col] = f2bf(v);
        else
          ((float*)Copaque)[(long)row * N + col] = v;
      }
    }
  }
}

// ---------------------------------------------------------------------------
// Fused flash-style merged attention (Q/K tiles staged via TDM).
//   blockIdx.y < M : scores = ([qg|qt].[kg|kt]^T)/8 (d=128) -> og
//   blockIdx.y == M: scores = (qt.kt^T)/8 (d=64)            -> ot
//   Grid: (L/64, M+1, B*H). Block: 128 threads = 4 waves, 16 q-rows each.
// ---------------------------------------------------------------------------
#define QK_STRIDE 136   // 128 + 8 pad ushorts (row = 68 DWORDs)
#define PV_STRIDE 72    // 64 + 8 pad
#define SEQ 512
#define DMODEL 512
#define NSER 4
// TDM padding: row data = 64 bf16 = 32 DWORDs (interval code 4 => 2^4*8B),
// pad to 68-DWORD LDS rows => 36 DWORDs pad (amount code 35 => 35+1).
#define TDM_IVL 4
#define TDM_PAD 35

__global__ __launch_bounds__(128) void flash_merge_kernel(
    const unsigned short* __restrict__ qt, const unsigned short* __restrict__ kt,
    const unsigned short* __restrict__ vt,
    const unsigned short* __restrict__ qg, const unsigned short* __restrict__ kg,
    const unsigned short* __restrict__ vg,
    float* __restrict__ ot, float* __restrict__ og) {
  __shared__ __align__(16) unsigned short Qs[64][QK_STRIDE];   // [qrow][d cat]
  __shared__ __align__(16) unsigned short Ks[64][QK_STRIDE];   // [key ][d cat]
  __shared__ __align__(16) unsigned short Vts[64][PV_STRIDE];  // [d][key] transposed
  __shared__ __align__(16) unsigned short Ps[64][PV_STRIDE];   // [qrow][key]

  const int tid  = threadIdx.x;
  const int wave = tid >> 5;
  const int lane = tid & 31;
  const int b    = blockIdx.z >> 3;
  const int h    = blockIdx.z & 7;
  const int m    = blockIdx.y;
  const bool timeStream = (m == NSER);
  const int q0   = blockIdx.x * 64;

  const long tbase = ((long)b * SEQ) * DMODEL + h * 64;
  const long gbase = timeStream ? 0
                   : ((long)((b * NSER + m) * SEQ)) * DMODEL + h * 64;

  const int nk = timeStream ? 2 : 4;     // d/32 k-steps for QK^T

  // ---- Q tile via TDM (wave 0 issues; completion covered by first k-block wait)
  if (wave == 0) {
    if (timeStream) {
      tdm_load_2d_bf16(lds_off(&Qs[0][0]), qt + tbase + (long)q0 * DMODEL,
                       64, 64, DMODEL, TDM_IVL, TDM_PAD);
    } else {
      tdm_load_2d_bf16(lds_off(&Qs[0][0]), qg + gbase + (long)q0 * DMODEL,
                       64, 64, DMODEL, TDM_IVL, TDM_PAD);
      tdm_load_2d_bf16(lds_off(&Qs[0][0]) + 128, qt + tbase + (long)q0 * DMODEL,
                       64, 64, DMODEL, TDM_IVL, TDM_PAD);
    }
  }

  float mrow[8], lrow[8];
#pragma unroll
  for (int r = 0; r < 8; ++r) { mrow[r] = -1e30f; lrow[r] = 0.f; }
  v8f o[4] = {vzero8(), vzero8(), vzero8(), vzero8()};

  const int hgrp = lane >> 4;
  const int cn   = lane & 15;

  for (int kb = 0; kb < SEQ / 64; ++kb) {
    const int k0 = kb * 64;
    __syncthreads();   // all waves done reading previous K/V tiles
    // ---- K tile via TDM (concat layout) ----
    if (wave == 0) {
      if (timeStream) {
        tdm_load_2d_bf16(lds_off(&Ks[0][0]), kt + tbase + (long)k0 * DMODEL,
                         64, 64, DMODEL, TDM_IVL, TDM_PAD);
      } else {
        tdm_load_2d_bf16(lds_off(&Ks[0][0]), kg + gbase + (long)k0 * DMODEL,
                         64, 64, DMODEL, TDM_IVL, TDM_PAD);
        tdm_load_2d_bf16(lds_off(&Ks[0][0]) + 128, kt + tbase + (long)k0 * DMODEL,
                         64, 64, DMODEL, TDM_IVL, TDM_PAD);
      }
    }
    // ---- V tile transposed (manual; TDM cannot transpose), overlaps TDM ----
    {
      const unsigned short* vsrc = timeStream ? (vt + tbase) : (vg + gbase);
      for (int idx = tid; idx < 4096; idx += 128) {
        const int key = idx >> 6;
        const int d   = idx & 63;
        Vts[d][key] = vsrc[(long)(k0 + key) * DMODEL + d];
      }
    }
    if (wave == 0) __builtin_amdgcn_s_wait_tensorcnt(0);
    __syncthreads();

    // ---- scores: 16 q-rows x 64 keys ----
    v8f s[4] = {vzero8(), vzero8(), vzero8(), vzero8()};
    for (int kk = 0; kk < nk; ++kk) {
      v16bf a = load_frag_a(&Qs[wave * 16][kk * 32], QK_STRIDE);
#pragma unroll
      for (int t = 0; t < 4; ++t) {
        v16bf bb = load_frag_b(&Ks[t * 16][kk * 32], QK_STRIDE);
        s[t] = wmma_bf16(a, bb, s[t]);
      }
    }

    // ---- online softmax update ----
#pragma unroll
    for (int r = 0; r < 8; ++r) {
      float mx = -1e30f;
#pragma unroll
      for (int t = 0; t < 4; ++t) {
        s[t][r] *= 0.125f;
        mx = fmaxf(mx, s[t][r]);
      }
      for (int off = 1; off < 16; off <<= 1)
        mx = fmaxf(mx, __shfl_xor(mx, off, 32));
      const float mnew  = fmaxf(mrow[r], mx);
      const float alpha = __expf(mrow[r] - mnew);
      float lsum = 0.f;
#pragma unroll
      for (int t = 0; t < 4; ++t) {
        const float p = __expf(s[t][r] - mnew);
        s[t][r] = p;
        lsum += p;
      }
      for (int off = 1; off < 16; off <<= 1)
        lsum += __shfl_xor(lsum, off, 32);
      lrow[r] = lrow[r] * alpha + lsum;
      mrow[r] = mnew;
#pragma unroll
      for (int t = 0; t < 4; ++t) o[t][r] *= alpha;
    }

    // ---- stage P (bf16) through LDS into A-fragment layout ----
#pragma unroll
    for (int t = 0; t < 4; ++t)
#pragma unroll
      for (int r = 0; r < 8; ++r)
        Ps[wave * 16 + hgrp * 8 + r][t * 16 + cn] = f2bf(s[t][r]);

    // ---- o += P(16x64) x V(64x64) ----
#pragma unroll
    for (int kk = 0; kk < 2; ++kk) {
      v16bf ap = load_frag_a(&Ps[wave * 16][kk * 32], PV_STRIDE);
#pragma unroll
      for (int t = 0; t < 4; ++t) {
        v16bf bv = load_frag_b(&Vts[t * 16][kk * 32], PV_STRIDE);
        o[t] = wmma_bf16(ap, bv, o[t]);
      }
    }
  }

  // ---- final normalize + store f32 ----
  float* dst = timeStream ? ot : og;
  const long obase = timeStream
      ? ((long)(b * SEQ + q0)) * DMODEL + h * 64
      : ((long)((b * NSER + m) * SEQ + q0)) * DMODEL + h * 64;
#pragma unroll
  for (int t = 0; t < 4; ++t)
#pragma unroll
    for (int r = 0; r < 8; ++r) {
      const int row = wave * 16 + hgrp * 8 + r;
      dst[obase + (long)row * DMODEL + t * 16 + cn] = o[t][r] / lrow[r];
    }
}

// ---------------------------------------------------------------------------
// Residual + LayerNorm over D=512:  out = LN(x + res) * g + b
// ---------------------------------------------------------------------------
__global__ __launch_bounds__(256) void ln_res_kernel(
    const float* __restrict__ x, const float* __restrict__ res,
    const float* __restrict__ g, const float* __restrict__ bb,
    float* __restrict__ out) {
  const int  t    = threadIdx.x;
  const long base = (long)blockIdx.x * 512;
  __shared__ float sm[8];

  float v0 = x[base + t]       + res[base + t];
  float v1 = x[base + t + 256] + res[base + t + 256];

  float s = v0 + v1;
  for (int off = 16; off; off >>= 1) s += __shfl_xor(s, off, 32);
  if ((t & 31) == 0) sm[t >> 5] = s;
  __syncthreads();
  float mean = 0.f;
#pragma unroll
  for (int i = 0; i < 8; ++i) mean += sm[i];
  mean *= (1.f / 512.f);
  __syncthreads();

  const float d0 = v0 - mean, d1 = v1 - mean;
  float vv = d0 * d0 + d1 * d1;
  for (int off = 16; off; off >>= 1) vv += __shfl_xor(vv, off, 32);
  if ((t & 31) == 0) sm[t >> 5] = vv;
  __syncthreads();
  float var = 0.f;
#pragma unroll
  for (int i = 0; i < 8; ++i) var += sm[i];
  var *= (1.f / 512.f);
  const float rstd = rsqrtf(var + 1e-5f);

  out[base + t]       = d0 * rstd * g[t]       + bb[t];
  out[base + t + 256] = d1 * rstd * g[t + 256] + bb[t + 256];
}

// ---------------------------------------------------------------------------
// Host launcher
// ---------------------------------------------------------------------------
extern "C" void kernel_launch(void* const* d_in, const int* in_sizes, int n_in,
                              void* d_out, int out_size, void* d_ws, size_t ws_size,
                              hipStream_t stream) {
  (void)in_sizes; (void)n_in; (void)out_size; (void)ws_size;

  const float* timeF = (const float*)d_in[0];
  const float* tgtF  = (const float*)d_in[1];
  const float* Wq_t = (const float*)d_in[2];  const float* bq_t = (const float*)d_in[3];
  const float* Wk_t = (const float*)d_in[4];  const float* bk_t = (const float*)d_in[5];
  const float* Wv_t = (const float*)d_in[6];  const float* bv_t = (const float*)d_in[7];
  const float* Wo_t = (const float*)d_in[8];  const float* bo_t = (const float*)d_in[9];
  const float* Wq_g = (const float*)d_in[10]; const float* bq_g = (const float*)d_in[11];
  const float* Wk_g = (const float*)d_in[12]; const float* bk_g = (const float*)d_in[13];
  const float* Wv_g = (const float*)d_in[14]; const float* bv_g = (const float*)d_in[15];
  const float* Wp   = (const float*)d_in[16]; const float* bp   = (const float*)d_in[17];
  const float* ln1_g = (const float*)d_in[18]; const float* ln1_b = (const float*)d_in[19];
  const float* ln2_g = (const float*)d_in[20]; const float* ln2_b = (const float*)d_in[21];
  const float* Wt1 = (const float*)d_in[22]; const float* bt1 = (const float*)d_in[23];
  const float* Wt2 = (const float*)d_in[24]; const float* bt2 = (const float*)d_in[25];
  const float* lnt_g = (const float*)d_in[26]; const float* lnt_b = (const float*)d_in[27];
  const float* Wg1 = (const float*)d_in[28]; const float* bg1 = (const float*)d_in[29];
  const float* Wg2 = (const float*)d_in[30]; const float* bg2 = (const float*)d_in[31];
  const float* lng_g = (const float*)d_in[32]; const float* lng_b = (const float*)d_in[33];

  // ---- workspace arena (peak ~164 MB, lifetime-based reuse) ----
  char* ws = (char*)d_ws;
  const long MB = 1l << 20;
  unsigned short* qt = (unsigned short*)(ws + 0 * MB);
  unsigned short* kt = (unsigned short*)(ws + 4 * MB);
  unsigned short* vt = (unsigned short*)(ws + 8 * MB);
  unsigned short* qg = (unsigned short*)(ws + 12 * MB);
  unsigned short* kg = (unsigned short*)(ws + 28 * MB);
  unsigned short* vg = (unsigned short*)(ws + 44 * MB);
  float* ot  = (float*)(ws + 60 * MB);
  float* og  = (float*)(ws + 68 * MB);
  float* tpr = (float*)(ws + 100 * MB);
  float* gpr = (float*)(ws + 108 * MB);
  float* TF  = (float*)(ws + 0 * MB);
  float* GF  = (float*)(ws + 12 * MB);
  unsigned short* TH = (unsigned short*)(ws + 44 * MB);
  float* TF2 = (float*)(ws + 60 * MB);
  unsigned short* GH = (unsigned short*)(ws + 68 * MB);
  float* GF2 = (float*)(ws + 132 * MB);

  float* tf_out = (float*)d_out;
  float* gf_out = (float*)d_out + 4096l * 512;

  const dim3 blk(256);
  // ---- QKV projections (f32 -> bf16), T x 512 x 512 ----
  gemm_wmma<false, true, false><<<dim3(4, 32),  blk, 0, stream>>>(timeF, Wq_t, bq_t, qt, 4096, 512, 512);
  gemm_wmma<false, true, false><<<dim3(4, 32),  blk, 0, stream>>>(timeF, Wk_t, bk_t, kt, 4096, 512, 512);
  gemm_wmma<false, true, false><<<dim3(4, 32),  blk, 0, stream>>>(timeF, Wv_t, bv_t, vt, 4096, 512, 512);
  gemm_wmma<false, true, false><<<dim3(4, 128), blk, 0, stream>>>(tgtF,  Wq_g, bq_g, qg, 16384, 512, 512);
  gemm_wmma<false, true, false><<<dim3(4, 128), blk, 0, stream>>>(tgtF,  Wk_g, bk_g, kg, 16384, 512, 512);
  gemm_wmma<false, true, false><<<dim3(4, 128), blk, 0, stream>>>(tgtF,  Wv_g, bv_g, vg, 16384, 512, 512);

  // ---- fused merged attention ----
  flash_merge_kernel<<<dim3(8, NSER + 1, 64), dim3(128), 0, stream>>>(
      qt, kt, vt, qg, kg, vg, ot, og);

  // ---- output projections ----
  gemm_wmma<false, false, false><<<dim3(4, 32),  blk, 0, stream>>>(ot, Wo_t, bo_t, tpr, 4096, 512, 512);
  gemm_wmma<false, false, false><<<dim3(4, 128), blk, 0, stream>>>(og, Wp,   bp,   gpr, 16384, 512, 512);

  // ---- residual + LN ----
  ln_res_kernel<<<4096,  blk, 0, stream>>>(tpr, timeF, ln1_g, ln1_b, TF);
  ln_res_kernel<<<16384, blk, 0, stream>>>(gpr, tgtF,  ln2_g, ln2_b, GF);

  // ---- time FFN ----
  gemm_wmma<false, true, true ><<<dim3(16, 32), blk, 0, stream>>>(TF, Wt1, bt1, TH, 4096, 2048, 512);
  gemm_wmma<true,  false, false><<<dim3(4, 32), blk, 0, stream>>>(TH, Wt2, bt2, TF2, 4096, 512, 2048);
  ln_res_kernel<<<4096, blk, 0, stream>>>(TF2, TF, lnt_g, lnt_b, tf_out);

  // ---- target FFN ----
  gemm_wmma<false, true, true ><<<dim3(16, 128), blk, 0, stream>>>(GF, Wg1, bg1, GH, 16384, 2048, 512);
  gemm_wmma<true,  false, false><<<dim3(4, 128), blk, 0, stream>>>(GH, Wg2, bg2, GF2, 16384, 512, 2048);
  ln_res_kernel<<<16384, blk, 0, stream>>>(GF2, GF, lng_g, lng_b, gf_out);
}
